// ParallelSelfAttention_27504970564335
// MI455X (gfx1250) — matrix operand order repaired and checked by reading, not compiled, not run on hardware
//
#include <hip/hip_runtime.h>

// ---------------------------------------------------------------------------
// Block-diagonal multi-head self-attention for gfx1250 (MI455X), all GEMMs on
// v_wmma_f32_16x16x32_f16.  fp32 inputs are converted once to f16 in d_ws.
//
// ws layout (halves): vh[8.4M] kh[8.4M] qh[8.4M] oh[8.4M] | mbuf/sbuf f32
// total ~68.2 MB.
//
// Round 2 changes:
//  - K1/K3: coalesced b128 weight reads + transpose-on-LDS-store (was 32
//    uncoalesced scalar global loads per thread per k-step)
//  - K1/K3: register-staged double-buffered LDS pipeline (1 barrier / k-step)
//  - K2b: global_prefetch of the next K/V chunk while current chunk computes
// ---------------------------------------------------------------------------

typedef _Float16 v16h __attribute__((ext_vector_type(16)));
typedef float    v8f  __attribute__((ext_vector_type(8)));
typedef unsigned int uv4 __attribute__((ext_vector_type(4)));
typedef float    fv4  __attribute__((ext_vector_type(4)));

#define WMMA(a, b, c) \
  __builtin_amdgcn_wmma_f32_16x16x32_f16(false, (a), false, (b), (short)0, (c), false, false)

// A-operand fragment: lane(row=ml) holds K = {hf*8..hf*8+7, 16+hf*8..16+hf*8+7}
__device__ __forceinline__ v16h frag_a(const _Float16* p) {
  union { v16h v; uv4 q[2]; } u;
  u.q[0] = *(const uv4*)(p);
  u.q[1] = *(const uv4*)(p + 16);
  return u.v;
}
// B-operand fragment: lane(col=ml) holds K = {hf*8..hf*8+7, 16+hf*8..16+hf*8+7}
__device__ __forceinline__ v16h frag_b(const _Float16* p) {
  union { v16h v; uv4 q[2]; } u;
  u.q[0] = *(const uv4*)(p);
  u.q[1] = *(const uv4*)(p + 16);
  return u.v;
}
__device__ __forceinline__ void cp16(_Float16* dst, const _Float16* src) {
  *(uv4*)dst = *(const uv4*)src;
}

// ---------------------------------------------------------------------------
// K1: QKV projections.  C[4096x512] = f16(X_block[4096x512]) @ f16(W_b) + bias
// grid (64 tiles, NB=4, proj=3), 256 thr = 8 waves, wave tile 64x64 (4x4 WMMA)
// Output in head-major f16: [n][b][h][s][64]
// ---------------------------------------------------------------------------
__global__ __launch_bounds__(256) void qkv_proj_kernel(
    const float* __restrict__ values, const float* __restrict__ keys,
    const float* __restrict__ query,
    const float* __restrict__ Wv, const float* __restrict__ Wk, const float* __restrict__ Wq,
    const float* __restrict__ bv, const float* __restrict__ bk, const float* __restrict__ bq,
    _Float16* __restrict__ vh, _Float16* __restrict__ kh, _Float16* __restrict__ qh)
{
  __shared__ __align__(16) _Float16 As[2][128 * 40];   // [row][k] 128x32 (+pad)
  __shared__ __align__(16) _Float16 Bs[2][256 * 40];   // B^T: [o][k] 256x32 (+pad)

  const int tid = threadIdx.x;
  const int wave = tid >> 5, lane = tid & 31;
  const int ml = lane & 15, hf = lane >> 4;
  const int tm = blockIdx.x >> 1, tn = blockIdx.x & 1;
  const int b = blockIdx.y, proj = blockIdx.z;

  const float* X    = proj == 0 ? values : proj == 1 ? keys : query;
  const float* W    = proj == 0 ? Wv : proj == 1 ? Wk : Wq;
  const float* bias = proj == 0 ? bv : proj == 1 ? bk : bq;
  _Float16*    dst  = proj == 0 ? vh : proj == 1 ? kh : qh;

  const int t0 = tm * 128, o0 = tn * 256;
  const int wm = (wave >> 2) * 64, wn = (wave & 3) * 64;

  const v8f vzero = {0.f, 0.f, 0.f, 0.f, 0.f, 0.f, 0.f, 0.f};
  v8f c[4][4];
  for (int i = 0; i < 4; i++)
    for (int j = 0; j < 4; j++) c[i][j] = vzero;

  fv4 aReg[4];   // A tile stage: 128x32 fp32, 4 float4 chunks / thread
  fv4 bReg[8];   // W tile stage: 32x256 fp32 row-major, 8 float4 / thread

  auto loadRegs = [&](int kb) {
    for (int i = 0; i < 4; i++) {
      int lin = tid + i * 256;                 // 1024 float4 chunks
      int row = lin >> 3, col = (lin & 7) * 4;
      aReg[i] = *(const fv4*)&X[(size_t)(t0 + row) * 2048 + b * 512 + kb + col];
    }
    for (int i = 0; i < 8; i++) {
      int lin = tid + i * 256;                 // 2048 float4 chunks, coalesced
      int k = lin >> 6, oc = (lin & 63) * 4;
      bReg[i] = *(const fv4*)&W[(size_t)b * 262144 + (size_t)(kb + k) * 512 + o0 + oc];
    }
  };
  auto storeLds = [&](int buf) {
    for (int i = 0; i < 4; i++) {
      int lin = tid + i * 256;
      int row = lin >> 3, col = (lin & 7) * 4;
      _Float16* p = &As[buf][row * 40 + col];
      p[0] = (_Float16)aReg[i][0]; p[1] = (_Float16)aReg[i][1];
      p[2] = (_Float16)aReg[i][2]; p[3] = (_Float16)aReg[i][3];
    }
    for (int i = 0; i < 8; i++) {              // transpose on store: Bs[o][k]
      int lin = tid + i * 256;
      int k = lin >> 6, oc = (lin & 63) * 4;
      for (int j = 0; j < 4; j++) Bs[buf][(oc + j) * 40 + k] = (_Float16)bReg[i][j];
    }
  };

  loadRegs(0);
  storeLds(0);
  __syncthreads();

  for (int kt = 0; kt < 16; kt++) {
    const int cur = kt & 1;
    if (kt < 15) loadRegs((kt + 1) * 32);      // overlap next tile with WMMAs
    const _Float16* Ab = As[cur];
    const _Float16* Bb = Bs[cur];
    v16h a[4], bb[4];
    for (int mi = 0; mi < 4; mi++) a[mi]  = frag_a(&Ab[(wm + mi * 16 + ml) * 40 + hf * 8]);
    for (int ni = 0; ni < 4; ni++) bb[ni] = frag_b(&Bb[(wn + ni * 16 + ml) * 40 + hf * 8]);
    for (int mi = 0; mi < 4; mi++)
      for (int ni = 0; ni < 4; ni++) c[mi][ni] = WMMA(a[mi], bb[ni], c[mi][ni]);
    if (kt < 15) storeLds(cur ^ 1);
    __syncthreads();
  }

  // store head-major f16 (C layout: M = vr + 8*hf, N = ml)
  for (int mi = 0; mi < 4; mi++)
    for (int ni = 0; ni < 4; ni++) {
      int oG = o0 + wn + ni * 16 + ml;
      float bs = bias[b * 512 + oG];
      int h = oG >> 6, d = oG & 63;
      for (int vr = 0; vr < 8; vr++) {
        int t = t0 + wm + mi * 16 + vr + 8 * hf;
        int n = t >> 10, s = t & 1023;
        size_t idx = ((((size_t)(n * 4 + b) * 8 + h) * 1024) + s) * 64 + d;
        dst[idx] = (_Float16)(c[mi][ni][vr] + bs);
      }
    }
}

// ---------------------------------------------------------------------------
// K2a: column softmax stats.  softmax is over the QUERY axis (axis=3 of
// nbhqk), so per key column l we need max_q and sum_q exp.  grid
// (lblock=8, b*h=32, n=4).  Wave owns 16 columns; C-layout puts one column
// per lane -> in-lane reduction + one shfl_xor(16) merge.
// ---------------------------------------------------------------------------
__global__ __launch_bounds__(256) void col_softmax_stats_kernel(
    const _Float16* __restrict__ qh, const _Float16* __restrict__ kh,
    const int* __restrict__ mask, float* __restrict__ mbuf, float* __restrict__ sbuf)
{
  __shared__ __align__(16) _Float16 Kt[128 * 72];   // [l][d]  (B^T for E)
  __shared__ __align__(16) _Float16 Qt[64 * 72];    // [q][d]  (A for E)

  const int tid = threadIdx.x, wave = tid >> 5, lane = tid & 31;
  const int ml = lane & 15, hf = lane >> 4;
  const int l0 = blockIdx.x * 128;
  const int nbh = blockIdx.z * 32 + blockIdx.y;
  const size_t headBase = (size_t)nbh * 1024;

  for (int i = 0; i < 4; i++) {
    int lin = tid + i * 256;                    // 8-half chunks, 1024 total
    int row = lin >> 3, c8 = (lin & 7) * 8;
    cp16(&Kt[row * 72 + c8], &kh[(headBase + l0 + row) * 64 + c8]);
  }
  __syncthreads();

  v16h bf0 = frag_b(&Kt[(wave * 16 + ml) * 72 + hf * 8]);
  v16h bf1 = frag_b(&Kt[(wave * 16 + ml) * 72 + 32 + hf * 8]);

  const int lG = l0 + wave * 16 + ml;
  const int* mcol = &mask[(size_t)blockIdx.z * 1048576 + lG];
  const v8f vzero = {0.f, 0.f, 0.f, 0.f, 0.f, 0.f, 0.f, 0.f};
  float runM = -3.0e38f, runS = 0.f;

  for (int qc = 0; qc < 16; qc++) {
    const int q0 = qc * 64;
    __syncthreads();
    for (int i = 0; i < 2; i++) {
      int lin = tid + i * 256;
      int row = lin >> 3, c8 = (lin & 7) * 8;
      cp16(&Qt[row * 72 + c8], &qh[(headBase + q0 + row) * 64 + c8]);
    }
    __syncthreads();
    if (qc < 15) {                               // pull next Q chunk toward WGP
      const _Float16* pq = &qh[(headBase + q0 + 64 + (tid >> 1)) * 64 + (tid & 1) * 32];
      __builtin_prefetch(pq, 0, 1);
    }

    float lg[4][8];
    for (int mi = 0; mi < 4; mi++) {
      v16h a0 = frag_a(&Qt[(mi * 16 + ml) * 72 + hf * 8]);
      v16h a1 = frag_a(&Qt[(mi * 16 + ml) * 72 + 32 + hf * 8]);
      v8f e = vzero;
      e = WMMA(a0, bf0, e);
      e = WMMA(a1, bf1, e);
      for (int vr = 0; vr < 8; vr++) {
        int q = q0 + mi * 16 + vr + 8 * hf;
        lg[mi][vr] = mcol[(size_t)q * 1024] ? e[vr] * 0.125f : -1.25e19f;
      }
    }
    float cm = -3.0e38f;
    for (int mi = 0; mi < 4; mi++)
      for (int vr = 0; vr < 8; vr++) cm = fmaxf(cm, lg[mi][vr]);
    float newM = fmaxf(runM, cm);
    float acc = runS * __expf(runM - newM);
    for (int mi = 0; mi < 4; mi++)
      for (int vr = 0; vr < 8; vr++) acc += __expf(lg[mi][vr] - newM);
    runS = acc; runM = newM;
  }
  // merge the two half-lanes that hold the same column (rows 0-7 vs 8-15)
  float oM = __shfl_xor(runM, 16);
  float oS = __shfl_xor(runS, 16);
  float M = fmaxf(runM, oM);
  float Ssum = runS * __expf(runM - M) + oS * __expf(oM - M);
  if (hf == 0) {
    mbuf[headBase + lG] = M;
    sbuf[headBase + lG] = Ssum;
  }
}

// ---------------------------------------------------------------------------
// K2b: O = softmax(E) @ V.  grid (qblock=8, b*h=32, n=4).  Wave owns 16 q
// rows; loop over 128-wide key chunks: E via WMMA, normalize with (m,s),
// route A through a per-wave LDS strip (C-layout -> A-layout), then
// O += A @ V with V transposed in LDS as B^T.  QA buffer aliases the Q tile
// (Q fragments are register-resident after the prologue).
// ---------------------------------------------------------------------------
__global__ __launch_bounds__(256) void attn_out_kernel(
    const _Float16* __restrict__ qh, const _Float16* __restrict__ kh,
    const _Float16* __restrict__ vh, const int* __restrict__ mask,
    const float* __restrict__ mbuf, const float* __restrict__ sbuf,
    _Float16* __restrict__ oh)
{
  __shared__ __align__(16) _Float16 Kt[128 * 72];   // [l][d]
  __shared__ __align__(16) _Float16 Vt[64 * 136];   // [d][l] (V^T = B^T)
  __shared__ __align__(16) _Float16 QA[128 * 72];   // Q tile, then A strips

  const int tid = threadIdx.x, wave = tid >> 5, lane = tid & 31;
  const int ml = lane & 15, hf = lane >> 4;
  const int q0 = blockIdx.x * 128;
  const int bh = blockIdx.y, n = blockIdx.z;
  const int nbh = n * 32 + bh;
  const int b = bh >> 3, h = bh & 7;
  const size_t headBase = (size_t)nbh * 1024;

  for (int i = 0; i < 4; i++) {
    int lin = tid + i * 256;
    int row = lin >> 3, c8 = (lin & 7) * 8;
    cp16(&QA[row * 72 + c8], &qh[(headBase + q0 + row) * 64 + c8]);
  }
  __syncthreads();
  v16h aq0 = frag_a(&QA[(wave * 16 + ml) * 72 + hf * 8]);
  v16h aq1 = frag_a(&QA[(wave * 16 + ml) * 72 + 32 + hf * 8]);
  __syncthreads();                                 // QA free for A strips now

  const v8f vzero = {0.f, 0.f, 0.f, 0.f, 0.f, 0.f, 0.f, 0.f};
  v8f oacc[4];
  for (int i = 0; i < 4; i++) oacc[i] = vzero;
  _Float16* Aw = &QA[wave * 1152];                 // 16 x 72 strip per wave

  for (int lc = 0; lc < 8; lc++) {
    const int l0 = lc * 128;
    __syncthreads();
    for (int i = 0; i < 4; i++) {
      int lin = tid + i * 256;
      int row = lin >> 3, c8 = (lin & 7) * 8;
      cp16(&Kt[row * 72 + c8], &kh[(headBase + l0 + row) * 64 + c8]);
    }
    for (int i = 0; i < 4; i++) {                  // V loaded transposed
      int lin = tid + i * 256;
      int row = lin >> 3, c8 = (lin & 7) * 8;
      union { uv4 q; _Float16 hl[8]; } u;
      u.q = *(const uv4*)&vh[(headBase + l0 + row) * 64 + c8];
      for (int j = 0; j < 8; j++) Vt[(c8 + j) * 136 + row] = u.hl[j];
    }
    __syncthreads();
    if (lc < 7) {                                  // prefetch next K/V chunk
      int row = tid >> 1, half64 = (tid & 1) * 32;
      const _Float16* pk = &kh[(headBase + l0 + 128 + row) * 64 + half64];
      const _Float16* pv = &vh[(headBase + l0 + 128 + row) * 64 + half64];
      __builtin_prefetch(pk, 0, 1);
      __builtin_prefetch(pv, 0, 1);
    }

    for (int half = 0; half < 2; half++) {         // 64 keys per half
      for (int ni = 0; ni < 4; ni++) {
        int nig = half * 4 + ni;
        v16h bk0 = frag_b(&Kt[(nig * 16 + ml) * 72 + hf * 8]);
        v16h bk1 = frag_b(&Kt[(nig * 16 + ml) * 72 + 32 + hf * 8]);
        v8f e = vzero;
        e = WMMA(aq0, bk0, e);
        e = WMMA(aq1, bk1, e);
        int lG = l0 + nig * 16 + ml;
        float mc  = mbuf[headBase + lG];
        float inv = 1.0f / sbuf[headBase + lG];
        const int* mcol = &mask[(size_t)n * 1048576 + lG];
        for (int vr = 0; vr < 8; vr++) {
          int q = q0 + wave * 16 + vr + 8 * hf;
          float lgt = mcol[(size_t)q * 1024] ? e[vr] * 0.125f : -1.25e19f;
          float a = __expf(lgt - mc) * inv;
          Aw[(vr + 8 * hf) * 72 + ni * 16 + ml] = (_Float16)a;
        }
      }
      __syncthreads();                             // A strips visible
      for (int kt = 0; kt < 2; kt++) {
        v16h aa = frag_a(&Aw[ml * 72 + kt * 32 + hf * 8]);
        for (int ni2 = 0; ni2 < 4; ni2++) {
          v16h bv2 = frag_b(&Vt[(ni2 * 16 + ml) * 136 + half * 64 + kt * 32 + hf * 8]);
          oacc[ni2] = WMMA(aa, bv2, oacc[ni2]);
        }
      }
      __syncthreads();                             // before Aw is rewritten
    }
  }
  // store O token-major f16: [(n*S+q)*NB + b][h*64+d]
  for (int ni2 = 0; ni2 < 4; ni2++) {
    int d = ni2 * 16 + ml;
    for (int vr = 0; vr < 8; vr++) {
      int q = q0 + wave * 16 + vr + 8 * hf;
      size_t idx = (((size_t)(n * 1024 + q) * 4 + b) * 512) + h * 64 + d;
      oh[idx] = (_Float16)oacc[ni2][vr];
    }
  }
}

// ---------------------------------------------------------------------------
// K3: output projection.  out[4096x512_b] = f16(O_b) @ f16(Wo_b) + bo, fp32.
// Same double-buffered pipeline as K1; A tile is already f16.
// ---------------------------------------------------------------------------
__global__ __launch_bounds__(256) void out_proj_kernel(
    const _Float16* __restrict__ oh, const float* __restrict__ Wo,
    const float* __restrict__ bo, float* __restrict__ out)
{
  __shared__ __align__(16) _Float16 As[2][128 * 40];
  __shared__ __align__(16) _Float16 Bs[2][256 * 40];

  const int tid = threadIdx.x;
  const int wave = tid >> 5, lane = tid & 31;
  const int ml = lane & 15, hf = lane >> 4;
  const int tm = blockIdx.x >> 1, tn = blockIdx.x & 1;
  const int b = blockIdx.y;
  const int t0 = tm * 128, o0 = tn * 256;
  const int wm = (wave >> 2) * 64, wn = (wave & 3) * 64;

  const v8f vzero = {0.f, 0.f, 0.f, 0.f, 0.f, 0.f, 0.f, 0.f};
  v8f c[4][4];
  for (int i = 0; i < 4; i++)
    for (int j = 0; j < 4; j++) c[i][j] = vzero;

  uv4 aReg[2];   // A tile stage: 128x32 f16, 2 8-half chunks / thread
  fv4 bReg[8];   // W tile stage: 32x256 fp32 row-major, coalesced

  auto loadRegs = [&](int kb) {
    for (int i = 0; i < 2; i++) {
      int lin = tid + i * 256;                 // 512 8-half chunks
      int row = lin >> 2, c8 = (lin & 3) * 8;
      aReg[i] = *(const uv4*)&oh[((size_t)(t0 + row) * 4 + b) * 512 + kb + c8];
    }
    for (int i = 0; i < 8; i++) {
      int lin = tid + i * 256;
      int k = lin >> 6, oc = (lin & 63) * 4;
      bReg[i] = *(const fv4*)&Wo[(size_t)b * 262144 + (size_t)(kb + k) * 512 + o0 + oc];
    }
  };
  auto storeLds = [&](int buf) {
    for (int i = 0; i < 2; i++) {
      int lin = tid + i * 256;
      int row = lin >> 2, c8 = (lin & 3) * 8;
      *(uv4*)&As[buf][row * 40 + c8] = aReg[i];
    }
    for (int i = 0; i < 8; i++) {
      int lin = tid + i * 256;
      int k = lin >> 6, oc = (lin & 63) * 4;
      for (int j = 0; j < 4; j++) Bs[buf][(oc + j) * 40 + k] = (_Float16)bReg[i][j];
    }
  };

  loadRegs(0);
  storeLds(0);
  __syncthreads();

  for (int kt = 0; kt < 16; kt++) {
    const int cur = kt & 1;
    if (kt < 15) loadRegs((kt + 1) * 32);
    const _Float16* Ab = As[cur];
    const _Float16* Bb = Bs[cur];
    v16h a[4], bb[4];
    for (int mi = 0; mi < 4; mi++) a[mi]  = frag_a(&Ab[(wm + mi * 16 + ml) * 40 + hf * 8]);
    for (int ni = 0; ni < 4; ni++) bb[ni] = frag_b(&Bb[(wn + ni * 16 + ml) * 40 + hf * 8]);
    for (int mi = 0; mi < 4; mi++)
      for (int ni = 0; ni < 4; ni++) c[mi][ni] = WMMA(a[mi], bb[ni], c[mi][ni]);
    if (kt < 15) storeLds(cur ^ 1);
    __syncthreads();
  }

  for (int mi = 0; mi < 4; mi++)
    for (int ni = 0; ni < 4; ni++) {
      int oG = o0 + wn + ni * 16 + ml;
      float bs = bo[b * 512 + oG];
      for (int vr = 0; vr < 8; vr++) {
        int t = t0 + wm + mi * 16 + vr + 8 * hf;
        out[(size_t)t * 2048 + b * 512 + oG] = c[mi][ni][vr] + bs;
      }
    }
}

// ---------------------------------------------------------------------------
extern "C" void kernel_launch(void* const* d_in, const int* in_sizes, int n_in,
                              void* d_out, int out_size, void* d_ws, size_t ws_size,
                              hipStream_t stream) {
  const float* values = (const float*)d_in[0];
  const float* keys   = (const float*)d_in[1];
  const float* query  = (const float*)d_in[2];
  const int*   mask   = (const int*)d_in[3];
  const float* Wv = (const float*)d_in[4];
  const float* bv = (const float*)d_in[5];
  const float* Wk = (const float*)d_in[6];
  const float* bk = (const float*)d_in[7];
  const float* Wq = (const float*)d_in[8];
  const float* bq = (const float*)d_in[9];
  const float* Wo = (const float*)d_in[10];
  const float* bo = (const float*)d_in[11];
  float* out = (float*)d_out;

  const size_t PLANE = 8388608;                   // N*S*E halves
  _Float16* vh = (_Float16*)d_ws;
  _Float16* kh = vh + PLANE;
  _Float16* qh = kh + PLANE;
  _Float16* oh = qh + PLANE;
  float* mbuf = (float*)(oh + PLANE);
  float* sbuf = mbuf + 131072;                    // N*NB*H*S

  qkv_proj_kernel<<<dim3(64, 4, 3), 256, 0, stream>>>(
      values, keys, query, Wv, Wk, Wq, bv, bk, bq, vh, kh, qh);
  col_softmax_stats_kernel<<<dim3(8, 32, 4), 256, 0, stream>>>(
      qh, kh, mask, mbuf, sbuf);
  attn_out_kernel<<<dim3(8, 32, 4), 256, 0, stream>>>(
      qh, kh, vh, mask, mbuf, sbuf, oh);
  out_proj_kernel<<<dim3(64, 4, 1), 256, 0, stream>>>(oh, Wo, bo, out);
}
